// GCN_9629316678024
// MI455X (gfx1250) — compile-verified, matrix operand order checked by reading, not run on hardware
//
#include <hip/hip_runtime.h>

// ---------------- problem constants ----------------
constexpr int GN      = 40000;     // nodes
constexpr int GE      = 640000;    // edges per adjacency
constexpr int NFEAT   = 256;
constexpr int NHID    = 128;
constexpr int NCAT    = 384;       // 3*NHID
constexpr int NCLASS  = 40;
constexpr float BN_EPS = 1e-5f;

// ---------------- WMMA vector types ----------------
typedef __attribute__((ext_vector_type(16))) __bf16 v16bf;
typedef __attribute__((ext_vector_type(8)))  __bf16 v8bf;
typedef __attribute__((ext_vector_type(8)))  float  v8f;

union V16U { v16bf v; v8bf h[2]; };

// Load a 16-element bf16 A/B fragment slice for one lane:
// 8 contiguous bf16 at p (K..K+7 for this lane-half) and 8 at p+16 (K+16..K+23).
__device__ __forceinline__ v16bf load_frag_bf16(const __bf16* p) {
  V16U u;
  u.h[0] = *(const v8bf*)(p);
  u.h[1] = *(const v8bf*)(p + 16);
  return u.v;
}

// Same slice but from an f32 matrix, with fused ReLU + cvt->bf16.
__device__ __forceinline__ v16bf load_frag_f32_relu(const float* p) {
  float4 f0 = ((const float4*)p)[0];
  float4 f1 = ((const float4*)p)[1];
  float4 f2 = ((const float4*)(p + 16))[0];
  float4 f3 = ((const float4*)(p + 16))[1];
  v16bf a;
  a[0]  = (__bf16)fmaxf(f0.x, 0.f);  a[1]  = (__bf16)fmaxf(f0.y, 0.f);
  a[2]  = (__bf16)fmaxf(f0.z, 0.f);  a[3]  = (__bf16)fmaxf(f0.w, 0.f);
  a[4]  = (__bf16)fmaxf(f1.x, 0.f);  a[5]  = (__bf16)fmaxf(f1.y, 0.f);
  a[6]  = (__bf16)fmaxf(f1.z, 0.f);  a[7]  = (__bf16)fmaxf(f1.w, 0.f);
  a[8]  = (__bf16)fmaxf(f2.x, 0.f);  a[9]  = (__bf16)fmaxf(f2.y, 0.f);
  a[10] = (__bf16)fmaxf(f2.z, 0.f);  a[11] = (__bf16)fmaxf(f2.w, 0.f);
  a[12] = (__bf16)fmaxf(f3.x, 0.f);  a[13] = (__bf16)fmaxf(f3.y, 0.f);
  a[14] = (__bf16)fmaxf(f3.z, 0.f);  a[15] = (__bf16)fmaxf(f3.w, 0.f);
  return a;
}

// ---------------- GEMM: C[M x 128] = act(A[M x K]) @ Bt[128 x K]^T (+bias) ----
// Bt is the weight pre-transposed to [out=128, in=K] row-major bf16.
// Each wave computes a 16-row x 64-col strip = 4 WMMA 16x16 tiles.
// blockDim = 256 (8 waves). grid = M/64 blocks (exact). K multiple of 32.
template <bool RELU_F32_A>
__global__ void __launch_bounds__(256)
gemm128_wmma(const void* __restrict__ Av, const __bf16* __restrict__ Bt,
             float* __restrict__ C, const float* __restrict__ bias,
             int M, int K) {
  const int tid  = threadIdx.x;
  const int wave = blockIdx.x * 8 + (tid >> 5);
  const int lane = tid & 31;
  const int mtile = wave >> 1;        // 2 strips of 64 cols per 16-row tile
  const int n64   = wave & 1;
  const int half  = lane >> 4;        // ISA lane-half (K phase)
  const int l15   = lane & 15;
  const int rowA  = mtile * 16 + l15;
  const int kOff  = half * 8;

  v8f acc0 = {}, acc1 = {}, acc2 = {}, acc3 = {};

  for (int kb = 0; kb < K; kb += 32) {
    v16bf a;
    if (RELU_F32_A)
      a = load_frag_f32_relu((const float*)Av + (size_t)rowA * K + kb + kOff);
    else
      a = load_frag_bf16((const __bf16*)Av + (size_t)rowA * K + kb + kOff);

    const __bf16* bbase = Bt + (size_t)(n64 * 64 + l15) * K + kb + kOff;
    v16bf b0 = load_frag_bf16(bbase);
    v16bf b1 = load_frag_bf16(bbase + (size_t)16 * K);
    v16bf b2 = load_frag_bf16(bbase + (size_t)32 * K);
    v16bf b3 = load_frag_bf16(bbase + (size_t)48 * K);

    acc0 = __builtin_amdgcn_wmma_f32_16x16x32_bf16(false, a, false, b0, (short)0, acc0, false, false);
    acc1 = __builtin_amdgcn_wmma_f32_16x16x32_bf16(false, a, false, b1, (short)0, acc1, false, false);
    acc2 = __builtin_amdgcn_wmma_f32_16x16x32_bf16(false, a, false, b2, (short)0, acc2, false, false);
    acc3 = __builtin_amdgcn_wmma_f32_16x16x32_bf16(false, a, false, b3, (short)0, acc3, false, false);
  }

  const int colBase = n64 * 64 + l15;
  float bv0 = 0.f, bv1 = 0.f, bv2 = 0.f, bv3 = 0.f;
  if (bias) {
    bv0 = bias[colBase];      bv1 = bias[colBase + 16];
    bv2 = bias[colBase + 32]; bv3 = bias[colBase + 48];
  }
  // C/D layout: VGPR r, lanes 0-15 -> row mtile*16+r, lanes 16-31 -> row +8.
  float* Crow = C + (size_t)(mtile * 16 + half * 8) * 128 + colBase;
#pragma unroll
  for (int r = 0; r < 8; ++r) {
    Crow[0]  = acc0[r] + bv0;
    Crow[16] = acc1[r] + bv1;
    Crow[32] = acc2[r] + bv2;
    Crow[48] = acc3[r] + bv3;
    Crow += 128;
  }
}

// ---------------- SpMM scatter: h[dst, band*128 + :] += w * sup[src, :] -------
// One wave per edge; 4 floats per lane; L2-resident atomics.
__global__ void __launch_bounds__(256)
spmm_scatter(const float* __restrict__ sup, const int* __restrict__ dst,
             const int* __restrict__ src, const float* __restrict__ w,
             float* __restrict__ h, int band, int E) {
  const int gw   = (blockIdx.x * blockDim.x + threadIdx.x) >> 5;
  const int lane = threadIdx.x & 31;
  if (gw >= E) return;
  const int   s  = src[gw];
  const int   d  = dst[gw];
  const float we = w[gw];
  float4 v = ((const float4*)(sup + (size_t)s * 128))[lane];
  float* hp = h + (size_t)d * NCAT + band * 128 + lane * 4;
  atomicAdd(hp + 0, v.x * we);
  atomicAdd(hp + 1, v.y * we);
  atomicAdd(hp + 2, v.z * we);
  atomicAdd(hp + 3, v.w * we);
}

// ---------------- elementwise helpers ----------------------------------------
__global__ void cvt_f32_bf16(const float* __restrict__ in, __bf16* __restrict__ out, int n) {
  for (int i = blockIdx.x * blockDim.x + threadIdx.x; i < n; i += gridDim.x * blockDim.x)
    out[i] = (__bf16)in[i];
}

// W [K x Nc] f32 row-major -> Wt [Nc x K] bf16 row-major
__global__ void wT_bf16(const float* __restrict__ W, __bf16* __restrict__ Wt, int K, int Nc) {
  int i = blockIdx.x * blockDim.x + threadIdx.x;
  if (i < K * Nc) {
    int k = i / Nc, n = i - k * Nc;
    Wt[(size_t)n * K + k] = (__bf16)W[i];
  }
}

__global__ void wT_f32(const float* __restrict__ W, float* __restrict__ Wt, int K, int Nc) {
  int i = blockIdx.x * blockDim.x + threadIdx.x;
  if (i < K * Nc) {
    int k = i / Nc, n = i - k * Nc;
    Wt[(size_t)n * K + k] = W[i];
  }
}

__global__ void init_h_bias(float* __restrict__ h, const float* __restrict__ b0,
                            const float* __restrict__ b1, const float* __restrict__ b2,
                            int Nrows) {
  int total = Nrows * NCAT;
  for (int i = blockIdx.x * blockDim.x + threadIdx.x; i < total; i += gridDim.x * blockDim.x) {
    int j = i % NCAT;
    int band = j >> 7, jj = j & 127;
    const float* b = (band == 0) ? b0 : (band == 1) ? b1 : b2;
    h[i] = b[jj];
  }
}

__global__ void stats_zero(float* __restrict__ stats) { stats[threadIdx.x] = 0.f; }

// per-column sum / sumsq over z[N x 128]; register partials then one atomic each
__global__ void __launch_bounds__(256)
bn_stats(const float* __restrict__ z, float* __restrict__ sums,
         float* __restrict__ sqs, int Nrows) {
  const int col = threadIdx.x & 127;
  const int sub = threadIdx.x >> 7;       // 0/1
  float s = 0.f, q = 0.f;
  for (int r = blockIdx.x * 2 + sub; r < Nrows; r += gridDim.x * 2) {
    float v = z[(size_t)r * 128 + col];
    s += v; q += v * v;
  }
  atomicAdd(&sums[col], s);
  atomicAdd(&sqs[col], q);
}

template <bool OUT_BF16>
__global__ void bn_apply(const float* __restrict__ z, const float* __restrict__ stats,
                         const float* __restrict__ g, const float* __restrict__ be,
                         void* __restrict__ out, int Nrows) {
  const float invN = 1.0f / (float)Nrows;
  int total = Nrows * 128;
  for (int i = blockIdx.x * blockDim.x + threadIdx.x; i < total; i += gridDim.x * blockDim.x) {
    int col = i & 127;
    float m   = stats[col] * invN;
    float var = stats[128 + col] * invN - m * m;
    float sc  = rsqrtf(var + BN_EPS) * g[col];
    float val = (z[i] - m) * sc + be[col];
    if (OUT_BF16) ((__bf16*)out)[i] = (__bf16)val;
    else          ((float*)out)[i]  = val;
  }
}

// ---------------- classifier: log_softmax(X @ Wm + bm) -----------------------
// Wmt is Wm pre-transposed to [40 x 128] f32. One wave per row; 8 rows/block.
__global__ void __launch_bounds__(256)
classifier_kernel(const float* __restrict__ X, const float* __restrict__ Wmt,
                  const float* __restrict__ bm, float* __restrict__ out, int Nrows) {
  __shared__ float logits[8][NCLASS];
  const int w    = threadIdx.x >> 5;
  const int lane = threadIdx.x & 31;
  const int row  = blockIdx.x * 8 + w;       // grid exact: Nrows % 8 == 0
  float4 xv = ((const float4*)(X + (size_t)row * 128))[lane];
  for (int c = 0; c < NCLASS; ++c) {
    float4 wv = ((const float4*)(Wmt + (size_t)c * 128))[lane];
    float p = xv.x * wv.x + xv.y * wv.y + xv.z * wv.z + xv.w * wv.w;
#pragma unroll
    for (int o = 16; o; o >>= 1) p += __shfl_xor(p, o, 32);
    if (lane == 0) logits[w][c] = p + bm[c];
  }
  __syncthreads();
  float v1 = logits[w][lane];                               // classes 0..31
  float v2 = (lane < 8) ? logits[w][32 + lane] : -3.0e38f;  // classes 32..39
  float mx = fmaxf(v1, v2);
#pragma unroll
  for (int o = 16; o; o >>= 1) mx = fmaxf(mx, __shfl_xor(mx, o, 32));
  float e = __expf(v1 - mx) + ((lane < 8) ? __expf(v2 - mx) : 0.f);
#pragma unroll
  for (int o = 16; o; o >>= 1) e += __shfl_xor(e, o, 32);
  float lse = __logf(e);
  out[(size_t)row * NCLASS + lane] = v1 - mx - lse;
  if (lane < 8) out[(size_t)row * NCLASS + 32 + lane] = v2 - mx - lse;
}

// ---------------- driver ------------------------------------------------------
extern "C" void kernel_launch(void* const* d_in, const int* in_sizes, int n_in,
                              void* d_out, int out_size, void* d_ws, size_t ws_size,
                              hipStream_t stream) {
  (void)in_sizes; (void)n_in; (void)out_size; (void)ws_size;
  const float* x   = (const float*)d_in[0];
  const int*   ei0 = (const int*)d_in[1];  const float* w0 = (const float*)d_in[2];
  const int*   ei1 = (const int*)d_in[3];  const float* w1 = (const float*)d_in[4];
  const int*   ei2 = (const int*)d_in[5];  const float* w2 = (const float*)d_in[6];
  const float* W11 = (const float*)d_in[7];  const float* b11 = (const float*)d_in[8];
  const float* W12 = (const float*)d_in[9];  const float* b12 = (const float*)d_in[10];
  const float* W13 = (const float*)d_in[11]; const float* b13 = (const float*)d_in[12];
  const float* Wf  = (const float*)d_in[13]; const float* bf  = (const float*)d_in[14];
  const float* g1  = (const float*)d_in[15]; const float* be1 = (const float*)d_in[16];
  const float* W21 = (const float*)d_in[17]; const float* b21 = (const float*)d_in[18];
  const float* W22 = (const float*)d_in[19]; const float* b22 = (const float*)d_in[20];
  const float* W23 = (const float*)d_in[21]; const float* b23 = (const float*)d_in[22];
  const float* Wf1 = (const float*)d_in[23]; const float* bf1 = (const float*)d_in[24];
  const float* g2  = (const float*)d_in[25]; const float* be2 = (const float*)d_in[26];
  const float* Wm  = (const float*)d_in[27]; const float* bm  = (const float*)d_in[28];
  float* out = (float*)d_out;

  // workspace carve-out (~113 MB)
  char* ws = (char*)d_ws;
  size_t off = 0;
  auto take = [&](size_t bytes) -> char* {
    char* p = ws + off;
    off += (bytes + 255) & ~(size_t)255;
    return p;
  };
  __bf16* xb    = (__bf16*)take((size_t)GN * NFEAT * 2);
  __bf16* W1t   = (__bf16*)take((size_t)3 * 128 * 256 * 2);
  __bf16* Wft   = (__bf16*)take((size_t)128 * 384 * 2);
  __bf16* W2t   = (__bf16*)take((size_t)3 * 128 * 128 * 2);
  __bf16* Wf1t  = (__bf16*)take((size_t)128 * 384 * 2);
  float*  Wmt   = (float*)take((size_t)NCLASS * 128 * 4);
  float*  stats = (float*)take(256 * 4);
  float*  sup   = (float*)take((size_t)GN * 128 * 4);   // per-conv GEMM out; reused as z
  float*  h     = (float*)take((size_t)GN * NCAT * 4);  // concat buffer; reused as bn2 (f32)
  __bf16* hbnb  = (__bf16*)take((size_t)GN * 128 * 2);  // layer-1 BN output (A of layer-2 GEMM)

  const int* eis[3] = {ei0, ei1, ei2};
  const float* ews[3] = {w0, w1, w2};

  // ---- precompute: bf16 x, transposed bf16 weights, transposed f32 Wm ----
  cvt_f32_bf16<<<4096, 256, 0, stream>>>(x, xb, GN * NFEAT);
  wT_bf16<<<(256 * 128 + 255) / 256, 256, 0, stream>>>(W11, W1t + (size_t)0 * 128 * 256, 256, 128);
  wT_bf16<<<(256 * 128 + 255) / 256, 256, 0, stream>>>(W12, W1t + (size_t)1 * 128 * 256, 256, 128);
  wT_bf16<<<(256 * 128 + 255) / 256, 256, 0, stream>>>(W13, W1t + (size_t)2 * 128 * 256, 256, 128);
  wT_bf16<<<(384 * 128 + 255) / 256, 256, 0, stream>>>(Wf, Wft, 384, 128);
  wT_bf16<<<(128 * 128 + 255) / 256, 256, 0, stream>>>(W21, W2t + (size_t)0 * 128 * 128, 128, 128);
  wT_bf16<<<(128 * 128 + 255) / 256, 256, 0, stream>>>(W22, W2t + (size_t)1 * 128 * 128, 128, 128);
  wT_bf16<<<(128 * 128 + 255) / 256, 256, 0, stream>>>(W23, W2t + (size_t)2 * 128 * 128, 128, 128);
  wT_bf16<<<(384 * 128 + 255) / 256, 256, 0, stream>>>(Wf1, Wf1t, 384, 128);
  wT_f32<<<(128 * 40 + 255) / 256, 256, 0, stream>>>(Wm, Wmt, 128, NCLASS);

  // ---- layer 1 ----
  init_h_bias<<<4096, 256, 0, stream>>>(h, b11, b12, b13, GN);
  for (int r = 0; r < 3; ++r) {
    gemm128_wmma<false><<<GN / 64, 256, 0, stream>>>(xb, W1t + (size_t)r * 128 * 256,
                                                     sup, nullptr, GN, 256);
    spmm_scatter<<<GE / 8, 256, 0, stream>>>(sup, eis[r], eis[r] + GE, ews[r], h, r, GE);
  }
  gemm128_wmma<true><<<GN / 64, 256, 0, stream>>>(h, Wft, sup, bf, GN, 384); // z1 in sup
  stats_zero<<<1, 256, 0, stream>>>(stats);
  bn_stats<<<512, 256, 0, stream>>>(sup, stats, stats + 128, GN);
  bn_apply<true><<<4096, 256, 0, stream>>>(sup, stats, g1, be1, hbnb, GN);

  // ---- layer 2 ----
  init_h_bias<<<4096, 256, 0, stream>>>(h, b21, b22, b23, GN);
  for (int r = 0; r < 3; ++r) {
    gemm128_wmma<false><<<GN / 64, 256, 0, stream>>>(hbnb, W2t + (size_t)r * 128 * 128,
                                                     sup, nullptr, GN, 128);
    spmm_scatter<<<GE / 8, 256, 0, stream>>>(sup, eis[r], eis[r] + GE, ews[r], h, r, GE);
  }
  gemm128_wmma<true><<<GN / 64, 256, 0, stream>>>(h, Wf1t, sup, bf1, GN, 384); // z2 in sup
  stats_zero<<<1, 256, 0, stream>>>(stats);
  bn_stats<<<512, 256, 0, stream>>>(sup, stats, stats + 128, GN);
  bn_apply<false><<<4096, 256, 0, stream>>>(sup, stats, g2, be2, (void*)h, GN); // bn2 f32 in h

  // ---- classifier + log_softmax ----
  classifier_kernel<<<GN / 8, 256, 0, stream>>>((const float*)h, Wmt, bm, out, GN);
}
